// Actor_60533269070268
// MI455X (gfx1250) — compile-verified, matrix operand order checked by reading, not run on hardware
//
#include <hip/hip_runtime.h>
#include <hip/hip_bf16.h>

// ---------------------------------------------------------------------------
// MI455X (gfx1250) actor-MLP forward.
// GEMMs run on v_wmma_f32_16x16x32_bf16 (wave32), double-buffered LDS tiles
// filled by GLOBAL_LOAD_ASYNC_TO_LDS_B128 (ASYNCcnt path), fp32 residual
// stream, fp32 reductions, fused epilogues.
// ---------------------------------------------------------------------------

typedef __attribute__((ext_vector_type(16))) __bf16 v16bf;
typedef __attribute__((ext_vector_type(8)))  float  v8f;

typedef int v4i __attribute__((vector_size(16)));
typedef __attribute__((address_space(1))) v4i v4i_as1;   // global
typedef __attribute__((address_space(3))) v4i v4i_as3;   // LDS

#if defined(__gfx1250__) && __has_builtin(__builtin_amdgcn_global_load_async_to_lds_b128)
#define USE_ASYNC 1
#else
#define USE_ASYNC 0
#endif

#define BATCH 16384
#define DIN   512
#define HDIM  1024
#define ADIM  128
#define NLAY  4

__device__ __forceinline__ unsigned short f2bf(float f) {
    unsigned int u = __float_as_uint(f);
    u += 0x7FFFu + ((u >> 16) & 1u);          // round-to-nearest-even
    return (unsigned short)(u >> 16);
}

// 16B global -> LDS copy: async on gfx1250, sync fallback otherwise
__device__ __forceinline__ void cp16(const unsigned short* g, unsigned short* l) {
#if USE_ASYNC
    __builtin_amdgcn_global_load_async_to_lds_b128((v4i_as1*)g, (v4i_as3*)l, 0, 0);
#else
    *reinterpret_cast<uint4*>(l) = *reinterpret_cast<const uint4*>(g);
#endif
}

__device__ __forceinline__ void wait_async() {
#if USE_ASYNC
#if __has_builtin(__builtin_amdgcn_s_wait_asynccnt)
    __builtin_amdgcn_s_wait_asynccnt(0);
#else
    asm volatile("s_wait_asynccnt 0x0" ::: "memory");
#endif
#endif
}

// ---------------------------------------------------------------------------
// rsnorm stage 1: per-column partial sum / sumsq over 256-row chunks.
// ---------------------------------------------------------------------------
__global__ __launch_bounds__(256) void rs_stage1(const float* __restrict__ state,
                                                 float* __restrict__ part) {
    const int b = blockIdx.x;           // 64 chunks of 256 rows
    const int t = threadIdx.x;          // 256 threads -> cols t and t+256
    float s0 = 0.f, q0 = 0.f, s1 = 0.f, q1 = 0.f;
    const float* p = state + (size_t)b * 256 * DIN;
    for (int r = 0; r < 256; ++r, p += DIN) {
        float a = p[t];       s0 += a; q0 += a * a;
        float c = p[t + 256]; s1 += c; q1 += c * c;
    }
    part[b * DIN + t]                  = s0;
    part[b * DIN + t + 256]            = s1;
    part[64 * DIN + b * DIN + t]       = q0;
    part[64 * DIN + b * DIN + t + 256] = q1;
}

// stage 2: finalize mu and rstd.  stats[c]=mu, stats[512+c]=1/sqrt(var+eps)
__global__ __launch_bounds__(256) void rs_stage2(const float* __restrict__ part,
                                                 float* __restrict__ stats) {
    const int c = blockIdx.x * 256 + threadIdx.x;   // 2 blocks -> 512 cols
    float s = 0.f, q = 0.f;
    for (int i = 0; i < 64; ++i) {
        s += part[i * DIN + c];
        q += part[64 * DIN + i * DIN + c];
    }
    const float inv = 1.0f / (float)BATCH;
    float mu  = s * inv;
    float var = fmaxf(q * inv - mu * mu, 0.001f);
    stats[c]       = mu;
    stats[DIN + c] = rsqrtf(var + 1e-5f);
}

// normalize + convert to bf16
__global__ __launch_bounds__(256) void rs_norm(const float* __restrict__ state,
                                               const float* __restrict__ stats,
                                               unsigned short* __restrict__ xn) {
    const int i = blockIdx.x * 256 + threadIdx.x;
    const int c = i & (DIN - 1);
    xn[i] = f2bf((state[i] - stats[c]) * stats[DIN + c]);
}

// ---------------------------------------------------------------------------
// Transpose-convert weights:  in f32 [K][N]  ->  out bf16 [N][K]
// grid (N/32, K/32), block (32,8)
// ---------------------------------------------------------------------------
__global__ __launch_bounds__(256) void tconv(const float* __restrict__ in,
                                             unsigned short* __restrict__ out,
                                             int K, int N) {
    __shared__ float tile[32][33];
    const int kb = blockIdx.y * 32, nb = blockIdx.x * 32;
    const int tx = threadIdx.x, ty = threadIdx.y;
    #pragma unroll
    for (int i = 0; i < 4; ++i) {
        int r = ty + i * 8;
        tile[r][tx] = in[(size_t)(kb + r) * N + nb + tx];
    }
    __syncthreads();
    #pragma unroll
    for (int i = 0; i < 4; ++i) {
        int r = ty + i * 8;
        out[(size_t)(nb + r) * K + kb + tx] = f2bf(tile[tx][r]);
    }
}

// ---------------------------------------------------------------------------
// Row LayerNorm over H=1024, output bf16:  (x-m)*rstd*g + b
// ---------------------------------------------------------------------------
__global__ __launch_bounds__(256) void ln_row(const float* __restrict__ x,
                                              const float* __restrict__ g,
                                              const float* __restrict__ b,
                                              unsigned short* __restrict__ out) {
    __shared__ float red1[256];
    __shared__ float red2[256];
    const int row = blockIdx.x, t = threadIdx.x;
    const float* xr = x + (size_t)row * HDIM;
    float v[4];
    float s = 0.f, q = 0.f;
    #pragma unroll
    for (int i = 0; i < 4; ++i) {
        v[i] = xr[t + i * 256];
        s += v[i]; q += v[i] * v[i];
    }
    red1[t] = s; red2[t] = q;
    __syncthreads();
    for (int off = 128; off > 0; off >>= 1) {
        if (t < off) { red1[t] += red1[t + off]; red2[t] += red2[t + off]; }
        __syncthreads();
    }
    const float m    = red1[0] * (1.0f / HDIM);
    const float var  = red2[0] * (1.0f / HDIM) - m * m;
    const float rstd = rsqrtf(var + 1e-5f);
    unsigned short* orow = out + (size_t)row * HDIM;
    #pragma unroll
    for (int i = 0; i < 4; ++i) {
        int c = t + i * 256;
        orow[c] = f2bf((v[i] - m) * rstd * g[c] + b[c]);
    }
}

// ---------------------------------------------------------------------------
// bf16 WMMA GEMM:  out[M,N] = epi( A[M,K] @ Bt[N,K]^T + bias[N] )
// Block tile 128x128, BK=32, double-buffered LDS (async fill on gfx1250).
// 256 threads = 8 waves; each wave 32x64 = 2x4 frags of wmma_f32_16x16x32_bf16.
// EPI: 0 = f32 +bias | 1 = bf16 relu(+bias) | 2 = f32 +bias+resid
//      3 = f32 clamp(+bias,-5,5) | 4 = f32 exp(clamp(+bias,-5,1))
// ---------------------------------------------------------------------------
#define LDA 40   // padded LDS row (elements); 80B keeps 16B alignment

template<int EPI>
__global__ __launch_bounds__(256) void gemm_wmma(
        const unsigned short* __restrict__ A,    // [M][K] bf16
        const unsigned short* __restrict__ Bt,   // [N][K] bf16
        const float* __restrict__ bias,          // [N]
        const float* __restrict__ resid,         // [M][N] or null
        void* __restrict__ out, int M, int N, int K) {
    __shared__ unsigned short As[2][128 * LDA];
    __shared__ unsigned short Bs[2][128 * LDA];

    const int tid  = threadIdx.x;
    const int wave = tid >> 5;
    const int lane = tid & 31;
    const int wm   = wave & 3;          // 4 waves along M
    const int wn   = wave >> 2;         // 2 waves along N
    const int lr   = lane & 15;
    const int khi  = lane >> 4;         // 0/1 -> K half select
    const int kh   = khi * 8;           // element offset within LDS row

    const int rowBase = blockIdx.y * 128;
    const int colBase = blockIdx.x * 128;

    // per-thread copy slots: 2x (A chunk + B chunk) of 16B
    const int cr0 = tid >> 2,           cc0 = (tid & 3) * 8;
    const int cr1 = (tid + 256) >> 2,   cc1 = ((tid + 256) & 3) * 8;

    auto stageFn = [&](int sbuf, int kc) {
        cp16(&A [(size_t)(rowBase + cr0) * K + kc + cc0], &As[sbuf][cr0 * LDA + cc0]);
        cp16(&Bt[(size_t)(colBase + cr0) * K + kc + cc0], &Bs[sbuf][cr0 * LDA + cc0]);
        cp16(&A [(size_t)(rowBase + cr1) * K + kc + cc1], &As[sbuf][cr1 * LDA + cc1]);
        cp16(&Bt[(size_t)(colBase + cr1) * K + kc + cc1], &Bs[sbuf][cr1 * LDA + cc1]);
    };

    const v8f zero = {0.f, 0.f, 0.f, 0.f, 0.f, 0.f, 0.f, 0.f};
    v8f acc[2][4];
    #pragma unroll
    for (int m = 0; m < 2; ++m)
        #pragma unroll
        for (int n = 0; n < 4; ++n) acc[m][n] = zero;

    stageFn(0, 0);
    wait_async();
    __syncthreads();

    int buf = 0;
    for (int kc = 0; kc < K; kc += 32) {
        if (kc + 32 < K) stageFn(buf ^ 1, kc + 32);   // prefetch next tile

        // fragments: lane<16 holds K{kh..kh+7, kh+16..kh+23} of its row
        v16bf af[2], bfr[4];
        #pragma unroll
        for (int m = 0; m < 2; ++m) {
            int off = (wm * 32 + m * 16 + lr) * LDA + kh;
            uint4* p = reinterpret_cast<uint4*>(&af[m]);
            p[0] = *reinterpret_cast<const uint4*>(&As[buf][off]);
            p[1] = *reinterpret_cast<const uint4*>(&As[buf][off + 16]);
        }
        #pragma unroll
        for (int n = 0; n < 4; ++n) {
            int off = (wn * 64 + n * 16 + lr) * LDA + kh;
            uint4* p = reinterpret_cast<uint4*>(&bfr[n]);
            p[0] = *reinterpret_cast<const uint4*>(&Bs[buf][off]);
            p[1] = *reinterpret_cast<const uint4*>(&Bs[buf][off + 16]);
        }
        #pragma unroll
        for (int m = 0; m < 2; ++m)
            #pragma unroll
            for (int n = 0; n < 4; ++n)
                acc[m][n] = __builtin_amdgcn_wmma_f32_16x16x32_bf16(
                    false, af[m], false, bfr[n], (short)0, acc[m][n], false, false);

        wait_async();
        __syncthreads();
        buf ^= 1;
    }

    // ---- fused epilogue (C/D layout: elem e -> M=e+8*khi, N=lr) ----
    float bv[4];
    #pragma unroll
    for (int n = 0; n < 4; ++n) bv[n] = bias[colBase + wn * 64 + n * 16 + lr];

    #pragma unroll
    for (int m = 0; m < 2; ++m) {
        #pragma unroll
        for (int n = 0; n < 4; ++n) {
            int col = colBase + wn * 64 + n * 16 + lr;
            #pragma unroll
            for (int e = 0; e < 8; ++e) {
                int row = rowBase + wm * 32 + m * 16 + khi * 8 + e;
                size_t oi = (size_t)row * N + col;
                float v = acc[m][n][e] + bv[n];
                if (EPI == 0) {
                    ((float*)out)[oi] = v;
                } else if (EPI == 1) {
                    ((unsigned short*)out)[oi] = f2bf(fmaxf(v, 0.f));
                } else if (EPI == 2) {
                    ((float*)out)[oi] = v + resid[oi];
                } else if (EPI == 3) {
                    ((float*)out)[oi] = fminf(fmaxf(v, -5.f), 5.f);
                } else {
                    ((float*)out)[oi] = __expf(fminf(fmaxf(v, -5.f), 1.f));
                }
            }
        }
    }
}

// ---------------------------------------------------------------------------
extern "C" void kernel_launch(void* const* d_in, const int* in_sizes, int n_in,
                              void* d_out, int out_size, void* d_ws, size_t ws_size,
                              hipStream_t stream) {
    (void)in_sizes; (void)n_in; (void)out_size; (void)ws_size;
    const float* state  = (const float*)d_in[0];
    const float* W_in   = (const float*)d_in[1];
    const float* b_in   = (const float*)d_in[2];
    const float* ln_g   = (const float*)d_in[3];
    const float* ln_b   = (const float*)d_in[4];
    const float* W1     = (const float*)d_in[5];
    const float* b1     = (const float*)d_in[6];
    const float* W2     = (const float*)d_in[7];
    const float* b2     = (const float*)d_in[8];
    const float* post_g = (const float*)d_in[9];
    const float* post_b = (const float*)d_in[10];
    const float* Wmu    = (const float*)d_in[11];
    const float* bmu    = (const float*)d_in[12];
    const float* Wstd   = (const float*)d_in[13];
    const float* bstd   = (const float*)d_in[14];

    // ---- workspace layout (bump allocator, 256B aligned) ----
    char* base = (char*)d_ws;
    size_t off = 0;
    auto alloc = [&](size_t bytes) {
        size_t o = off; off = (off + bytes + 255) & ~(size_t)255; return (void*)(base + o);
    };
    float*          part   = (float*)         alloc(2ull * 64 * DIN * 4);
    float*          stats  = (float*)         alloc(2ull * DIN * 4);
    unsigned short* xn     = (unsigned short*)alloc((size_t)BATCH * DIN * 2);
    unsigned short* wt_in  = (unsigned short*)alloc((size_t)HDIM * DIN * 2);
    unsigned short* wt1    = (unsigned short*)alloc((size_t)NLAY * HDIM * HDIM * 2);
    unsigned short* wt2    = (unsigned short*)alloc((size_t)NLAY * HDIM * HDIM * 2);
    unsigned short* wt_mu  = (unsigned short*)alloc((size_t)ADIM * HDIM * 2);
    unsigned short* wt_std = (unsigned short*)alloc((size_t)ADIM * HDIM * 2);
    float*          x      = (float*)         alloc((size_t)BATCH * HDIM * 4);
    unsigned short* h      = (unsigned short*)alloc((size_t)BATCH * HDIM * 2);
    unsigned short* t      = (unsigned short*)alloc((size_t)BATCH * HDIM * 2);

    // ---- rsnorm: batch stats + normalize to bf16 ----
    rs_stage1<<<64, 256, 0, stream>>>(state, part);
    rs_stage2<<<2, 256, 0, stream>>>(part, stats);
    rs_norm<<<(BATCH * DIN) / 256, 256, 0, stream>>>(state, stats, xn);

    // ---- weight transpose-convert to bf16 [N][K] ----
    tconv<<<dim3(HDIM / 32, DIN / 32), dim3(32, 8), 0, stream>>>(W_in, wt_in, DIN, HDIM);
    for (int l = 0; l < NLAY; ++l) {
        tconv<<<dim3(HDIM / 32, HDIM / 32), dim3(32, 8), 0, stream>>>(
            W1 + (size_t)l * HDIM * HDIM, wt1 + (size_t)l * HDIM * HDIM, HDIM, HDIM);
        tconv<<<dim3(HDIM / 32, HDIM / 32), dim3(32, 8), 0, stream>>>(
            W2 + (size_t)l * HDIM * HDIM, wt2 + (size_t)l * HDIM * HDIM, HDIM, HDIM);
    }
    tconv<<<dim3(ADIM / 32, HDIM / 32), dim3(32, 8), 0, stream>>>(Wmu,  wt_mu,  HDIM, ADIM);
    tconv<<<dim3(ADIM / 32, HDIM / 32), dim3(32, 8), 0, stream>>>(Wstd, wt_std, HDIM, ADIM);

    // ---- input projection ----
    gemm_wmma<0><<<dim3(HDIM / 128, BATCH / 128), 256, 0, stream>>>(
        xn, wt_in, b_in, nullptr, x, BATCH, HDIM, DIN);

    // ---- residual MLP blocks ----
    for (int l = 0; l < NLAY; ++l) {
        ln_row<<<BATCH, 256, 0, stream>>>(x, ln_g + l * HDIM, ln_b + l * HDIM, h);
        gemm_wmma<1><<<dim3(HDIM / 128, BATCH / 128), 256, 0, stream>>>(
            h, wt1 + (size_t)l * HDIM * HDIM, b1 + l * HDIM, nullptr, t, BATCH, HDIM, HDIM);
        gemm_wmma<2><<<dim3(HDIM / 128, BATCH / 128), 256, 0, stream>>>(
            t, wt2 + (size_t)l * HDIM * HDIM, b2 + l * HDIM, x, x, BATCH, HDIM, HDIM);
    }

    // ---- post LN + heads ----
    ln_row<<<BATCH, 256, 0, stream>>>(x, post_g, post_b, h);
    float* out_mu  = (float*)d_out;
    float* out_std = out_mu + (size_t)BATCH * ADIM;
    gemm_wmma<3><<<dim3(ADIM / 128, BATCH / 128), 256, 0, stream>>>(
        h, wt_mu, bmu, nullptr, out_mu, BATCH, ADIM, HDIM);
    gemm_wmma<4><<<dim3(ADIM / 128, BATCH / 128), 256, 0, stream>>>(
        h, wt_std, bstd, nullptr, out_std, BATCH, ADIM, HDIM);
}
